// GAT_36945308680797
// MI455X (gfx1250) — matrix operand order repaired and checked by reading, not compiled, not run on hardware
//
#include <hip/hip_runtime.h>
#include <hip/hip_bf16.h>

// ---------------------------------------------------------------------------
// Problem constants (match reference)
// ---------------------------------------------------------------------------
#define Ndim 4096
#define FIN  512
#define NH   4
#define OD   256
#define HOD  1024          // NH * OD
#define ALPHA_SLOPE 0.2f
#define NEGV (-9e15f)

// ---------------------------------------------------------------------------
// WMMA types (CDNA5 gfx1250, wave32)
// ---------------------------------------------------------------------------
typedef __attribute__((ext_vector_type(16))) __bf16        v16bf;
typedef __attribute__((ext_vector_type(8)))  float         v8f;
typedef __attribute__((ext_vector_type(4)))  unsigned int  v4u;   // trivial 16B vec

union BF16Frag {
    unsigned short us[16];
    v4u  q[2];
    v16bf v;
};

__device__ __forceinline__ unsigned short f2bf(float f) {
    unsigned int u = __float_as_uint(f);
    unsigned int r = u + 0x7FFFu + ((u >> 16) & 1u);   // round-to-nearest-even
    return (unsigned short)(r >> 16);
}

// Fragment-order LDS slot mapping (ISA 7.12.2, bf16 16x16x32):
//  A (16x32, m x k): lane = (m&15) | (k&8 ? 16 : 0); slot = (k&7) | (k&16 ? 8 : 0)
//  B (32x16, k x n): lane = (n&15) | (k&8 ? 16 : 0); slot = (k&7) | (k&16 ? 8 : 0)
// Each lane's 16 slots are 32 contiguous bytes -> 2x ds_load_b128.
__device__ __forceinline__ void load_frag(BF16Frag& f,
                                          const unsigned short* __restrict__ p) {
    const v4u* q = (const v4u*)p;
    f.q[0] = q[0];
    f.q[1] = q[1];
}

// async 16-byte global->LDS copy (GLOBAL_LOAD_ASYNC_TO_LDS_B128, ASYNCcnt)
__device__ __forceinline__ void async_copy_b128(unsigned lds_off, unsigned goff,
                                                unsigned long long gbase) {
    asm volatile("global_load_async_to_lds_b128 %0, %1, %2"
                 :: "v"(lds_off), "v"(goff), "s"(gbase)
                 : "memory");
}
__device__ __forceinline__ void wait_async_all() {
    asm volatile("s_wait_asynccnt 0x0" ::: "memory");
}

// ---------------------------------------------------------------------------
// Kernel 1: per-head feature GEMM  Y[h] = X @ W[h]
//   block = 128 thr (4 waves); block tile 64(M) x 64(N); K-step 32 (bf16 WMMA)
//   wave w owns rows [16w,16w+16); 4 B-frags cover the 64 N columns
// ---------------------------------------------------------------------------
__global__ __launch_bounds__(128) void gemm_bf16_wmma(
    const float* __restrict__ X, const float* __restrict__ W,
    float* __restrict__ Y, int Nrows, int F, int Ocols)
{
    const int h  = blockIdx.z;
    const int m0 = blockIdx.x * 64;
    const int n0 = blockIdx.y * 64;
    const float* Wh = W + (size_t)h * F * Ocols;
    float*       Yh = Y + (size_t)h * Nrows * Ocols;

    __shared__ __align__(16) unsigned short Afrag[4][32][16];  // 4 KB
    __shared__ __align__(16) unsigned short Bfrag[4][32][16];  // 4 KB

    const int tid = threadIdx.x, wave = tid >> 5, lane = tid & 31;
    v8f acc[4] = {};

    for (int k0 = 0; k0 < F; k0 += 32) {
        // stage A tile 64x32 into fragment order (16 elems/thread)
#pragma unroll
        for (int i = 0; i < 16; ++i) {
            int idx = tid + i * 128;
            int r = idx >> 5, c = idx & 31;                    // r=m, c=k
            Afrag[r >> 4][(r & 15) | ((c & 8) ? 16 : 0)]
                 [(c & 7) | ((c & 16) ? 8 : 0)] =
                f2bf(X[(size_t)(m0 + r) * F + k0 + c]);
        }
        // stage B tile 32x64 into fragment order (16 elems/thread)
#pragma unroll
        for (int i = 0; i < 16; ++i) {
            int idx = tid + i * 128;
            int r = idx >> 6, c = idx & 63;                    // r=k, c=n
            Bfrag[c >> 4][(c & 15) | ((r & 8) ? 16 : 0)][(r & 7) | ((r & 16) ? 8 : 0)] =
                f2bf(Wh[(size_t)(k0 + r) * Ocols + n0 + c]);
        }
        __syncthreads();

        BF16Frag a;
        load_frag(a, &Afrag[wave][lane][0]);
#pragma unroll
        for (int f = 0; f < 4; ++f) {
            BF16Frag b;
            load_frag(b, &Bfrag[f][lane][0]);
            acc[f] = __builtin_amdgcn_wmma_f32_16x16x32_bf16(
                false, a.v, false, b.v, (short)0, acc[f], false, false);
        }
        __syncthreads();
    }

    const int mo = (lane >> 4) * 8;
    const int nc = lane & 15;
#pragma unroll
    for (int f = 0; f < 4; ++f)
#pragma unroll
        for (int r = 0; r < 8; ++r)
            Yh[(size_t)(m0 + wave * 16 + r + mo) * Ocols + n0 + f * 16 + nc] =
                acc[f][r];
}

// ---------------------------------------------------------------------------
// Kernel 2: attention coefficients  fs[h,n]=h[h,n,:].a_s[h]  fn likewise
// ---------------------------------------------------------------------------
__global__ __launch_bounds__(256) void attn_coeff(
    const float* __restrict__ Hmat, const float* __restrict__ a_s,
    const float* __restrict__ a_n, float* __restrict__ fs,
    float* __restrict__ fn)
{
    int idx = blockIdx.x * 256 + threadIdx.x;          // 0 .. NH*Ndim-1
    int h = idx / Ndim;
    const float* row = Hmat + (size_t)idx * OD;
    const float* as = a_s + h * OD;
    const float* an = a_n + h * OD;
    float s = 0.f, t = 0.f;
#pragma unroll 4
    for (int o = 0; o < OD; ++o) {
        float v = row[o];
        s += v * as[o];
        t += v * an[o];
    }
    fs[idx] = s;
    fn[idx] = t;
}

// ---------------------------------------------------------------------------
// Kernel 3: fused flash-attention GAT layer
//   e = leakyrelu((fs_i + fn_j) * M_ij); masked by adj; online softmax;
//   out = softmax(e) @ h   (bf16 WMMA, fp32 accum/stats)
//   grid (Ndim/16, NH); block 128 = 4 waves; wave w owns O-slice [64w,64w+64)
//   mode 0: concat + ELU  -> Y[N, HOD];  mode 1: per-head -> Y[NH,N,OD]
// ---------------------------------------------------------------------------
__global__ __launch_bounds__(128) void gat_attention(
    const float* __restrict__ Hmat,   // [NH, Ndim, OD]
    const float* __restrict__ fs, const float* __restrict__ fn,  // [NH, Ndim]
    const float* __restrict__ adj, const float* __restrict__ Mprox, // [N,N]
    float* __restrict__ Y, int mode)
{
    const int h  = blockIdx.y;
    const int i0 = blockIdx.x * 16;
    const int tid = threadIdx.x, wave = tid >> 5, lane = tid & 31;

    __shared__ float fsr[16];
    __shared__ float et[16][32];
    __shared__ float rowm[16], rowl[16], rowsc[16];
    __shared__ __align__(16) unsigned short Pfrag[32][16];      // probs, 1 KB
    __shared__ __align__(16) unsigned short Hfrag[16][32][16];  // h tile, 16 KB

    if (tid < 16) {
        fsr[tid]  = fs[h * Ndim + i0 + tid];
        rowm[tid] = -3.0e38f;
        rowl[tid] = 0.f;
    }
    v8f acc[4] = {};
    const float* Hh = Hmat + (size_t)h * Ndim * OD;
    __syncthreads();

    for (int j0 = 0; j0 < Ndim; j0 += 32) {
        // stage 32(K) x 256(O) h-tile (L2-resident) in fragment order
#pragma unroll 8
        for (int i = 0; i < 64; ++i) {
            int idx = tid + i * 128;
            int r = idx >> 8, c = idx & 255;                   // r=k, c=o
            Hfrag[c >> 4][(c & 15) | ((r & 8) ? 16 : 0)][(r & 7) | ((r & 16) ? 8 : 0)] =
                f2bf(Hh[(size_t)(j0 + r) * OD + c]);
        }
        // logits tile 16x32: modulate by M, leaky-relu, adjacency mask
#pragma unroll
        for (int i = 0; i < 4; ++i) {
            int idx = tid + i * 128;
            int r = idx >> 5, c = idx & 31;
            int j = j0 + c;
            float ev = fsr[r] + fn[h * Ndim + j];
            ev *= Mprox[(size_t)(i0 + r) * Ndim + j];
            ev = (ev > 0.f) ? ev : ALPHA_SLOPE * ev;
            ev = (adj[(size_t)(i0 + r) * Ndim + j] > 0.f) ? ev : NEGV;
            et[r][c] = ev;
        }
        __syncthreads();

        // online softmax statistics (one thread per query row)
        if (tid < 16) {
            float mo = rowm[tid], mn = mo;
#pragma unroll
            for (int c = 0; c < 32; ++c) mn = fmaxf(mn, et[tid][c]);
            float sc = __expf(mo - mn);
            float l  = rowl[tid] * sc;
#pragma unroll
            for (int c = 0; c < 32; ++c) {
                float p = __expf(et[tid][c] - mn);
                l += p;
                Pfrag[tid | ((c & 8) ? 16 : 0)]
                     [(c & 7) | ((c & 16) ? 8 : 0)] = f2bf(p);
            }
            rowm[tid] = mn; rowl[tid] = l; rowsc[tid] = sc;
        }
        __syncthreads();

        // rescale running accumulators by exp(m_old - m_new)
        const int mo = (lane >> 4) * 8;
#pragma unroll
        for (int f = 0; f < 4; ++f)
#pragma unroll
            for (int r = 0; r < 8; ++r)
                acc[f][r] *= rowsc[r + mo];

        // P (16x32) as A, h-tile columns as B; 4 WMMAs cover this wave's 64 O
        BF16Frag a;
        load_frag(a, &Pfrag[lane][0]);
#pragma unroll
        for (int f = 0; f < 4; ++f) {
            BF16Frag b;
            load_frag(b, &Hfrag[wave * 4 + f][lane][0]);
            acc[f] = __builtin_amdgcn_wmma_f32_16x16x32_bf16(
                false, a.v, false, b.v, (short)0, acc[f], false, false);
        }
        __syncthreads();
    }

    // epilogue: divide by softmax denominator; concat+ELU or per-head store
    const int mo = (lane >> 4) * 8;
    const int nc = lane & 15;
#pragma unroll
    for (int f = 0; f < 4; ++f) {
#pragma unroll
        for (int r = 0; r < 8; ++r) {
            int m   = r + mo;
            int row = i0 + m;
            int o   = wave * 64 + f * 16 + nc;
            float v = acc[f][r] / rowl[m];
            if (mode == 0) {
                v = (v > 0.f) ? v : (__expf(v) - 1.f);          // ELU
                Y[(size_t)row * HOD + h * OD + o] = v;
            } else {
                Y[((size_t)h * Ndim + row) * OD + o] = v;
            }
        }
    }
}

// ---------------------------------------------------------------------------
// Kernel 4: head-mean + row L2 normalize; emit z (f32) and z (bf16)
// ---------------------------------------------------------------------------
__global__ __launch_bounds__(256) void mean_normalize(
    const float* __restrict__ Yh,       // [NH, Ndim, OD]
    float* __restrict__ zout,           // [Ndim, OD]
    unsigned short* __restrict__ zbf)   // [Ndim, OD] bf16
{
    const int row = blockIdx.x;
    const int o   = threadIdx.x;
    float v = 0.f;
#pragma unroll
    for (int h = 0; h < NH; ++h)
        v += Yh[((size_t)h * Ndim + row) * OD + o];
    v *= (1.f / NH);

    __shared__ float red[256];
    red[o] = v * v;
    __syncthreads();
    for (int s = 128; s > 0; s >>= 1) {
        if (o < s) red[o] += red[o + s];
        __syncthreads();
    }
    float nrm = fmaxf(sqrtf(red[0]), 1e-12f);
    float z = v / nrm;
    zout[(size_t)row * OD + o] = z;
    zbf[(size_t)row * OD + o]  = f2bf(z);
}

// ---------------------------------------------------------------------------
// Kernel 5: decode  A_pred = sigmoid(z z^T)  via bf16 WMMA
//   block 128 thr = 4 waves, tile 64(i) x 16(j); full K=256 staged in LDS
//   in fragment order by GLOBAL_LOAD_ASYNC_TO_LDS_B128 (ASYNCcnt-tracked)
// ---------------------------------------------------------------------------
__global__ __launch_bounds__(128) void decode_sigmoid(
    const unsigned short* __restrict__ zbf, float* __restrict__ Apred)
{
    const int i0 = blockIdx.x * 64;
    const int j0 = blockIdx.y * 16;
    const int tid = threadIdx.x, wave = tid >> 5, lane = tid & 31;

    __shared__ __align__(16) unsigned short Afrag[4][8][32][16]; // 32 KB
    __shared__ __align__(16) unsigned short Bfrag[8][32][16];    //  8 KB

    const unsigned aBase = (unsigned)(size_t)&Afrag[0][0][0][0];
    const unsigned bBase = (unsigned)(size_t)&Bfrag[0][0][0];
    const unsigned long long gz = (unsigned long long)zbf;

    // A rows i0..i0+63: 2048 16-byte chunks, each a contiguous global run
    //   chunk(rb,kc,lane,half) <-> z[rb*16+(lane&15)][kc*32 + 8*(lane>=16) + 16*half ..+8]
#pragma unroll
    for (int i = 0; i < 16; ++i) {
        int id   = tid + i * 128;
        int half = id & 1, ln = (id >> 1) & 31, kc = (id >> 6) & 7, rb = id >> 9;
        int row  = i0 + rb * 16 + (ln & 15);
        int kh   = (ln >> 4) * 8;
        unsigned goff = (unsigned)((row * OD + kc * 32 + kh + half * 16) * 2);
        unsigned loff = aBase +
            (unsigned)(((((rb * 8 + kc) * 32) + ln) * 16 + half * 8) * 2);
        async_copy_b128(loff, goff, gz);
    }
    // B rows j0..j0+15: 512 chunks
#pragma unroll
    for (int i = 0; i < 4; ++i) {
        int id   = tid + i * 128;
        int half = id & 1, ln = (id >> 1) & 31, kc = id >> 6;
        int row  = j0 + (ln & 15);
        int ko   = (ln >> 4) * 8;
        unsigned goff = (unsigned)((row * OD + kc * 32 + ko + half * 16) * 2);
        unsigned loff = bBase + (unsigned)(((kc * 32 + ln) * 16 + half * 8) * 2);
        async_copy_b128(loff, goff, gz);
    }
    wait_async_all();
    __syncthreads();

    v8f acc = {};
#pragma unroll
    for (int kc = 0; kc < 8; ++kc) {
        BF16Frag a, b;
        load_frag(a, &Afrag[wave][kc][lane][0]);
        load_frag(b, &Bfrag[kc][lane][0]);
        acc = __builtin_amdgcn_wmma_f32_16x16x32_bf16(
            false, a.v, false, b.v, (short)0, acc, false, false);
    }

    const int mo = (lane >> 4) * 8;
    const int nc = lane & 15;
#pragma unroll
    for (int r = 0; r < 8; ++r) {
        float v = acc[r];
        v = 1.f / (1.f + __expf(-v));
        Apred[(size_t)(i0 + wave * 16 + r + mo) * Ndim + (j0 + nc)] = v;
    }
}

// ---------------------------------------------------------------------------
// Host launcher
// ---------------------------------------------------------------------------
extern "C" void kernel_launch(void* const* d_in, const int* in_sizes, int n_in,
                              void* d_out, int out_size, void* d_ws, size_t ws_size,
                              hipStream_t stream) {
    const float* x     = (const float*)d_in[0];
    const float* adj   = (const float*)d_in[1];
    const float* Mprox = (const float*)d_in[2];
    const float* W0    = (const float*)d_in[3];
    const float* as0   = (const float*)d_in[4];
    const float* an0   = (const float*)d_in[5];
    const float* W1    = (const float*)d_in[6];
    const float* as1   = (const float*)d_in[7];
    const float* an1   = (const float*)d_in[8];

    float* out = (float*)d_out;                 // [N*N] A_pred ++ [N*OD] z
    float* ws  = (float*)d_ws;

    size_t off = 0;
    float* h0   = ws + off; off += (size_t)NH * Ndim * OD;   // 16 MB
    float* fs0  = ws + off; off += (size_t)NH * Ndim;
    float* fn0  = ws + off; off += (size_t)NH * Ndim;
    float* out0 = ws + off; off += (size_t)Ndim * HOD;       // 16 MB
    float* h1   = ws + off; off += (size_t)NH * Ndim * OD;   // 16 MB
    float* fs1  = ws + off; off += (size_t)NH * Ndim;
    float* fn1  = ws + off; off += (size_t)NH * Ndim;
    float* yh1  = ws + off; off += (size_t)NH * Ndim * OD;   // 16 MB
    unsigned short* zbf = (unsigned short*)(ws + off);       //  2 MB

    float* zout = out + (size_t)Ndim * Ndim;

    // ---- layer 0 ----
    gemm_bf16_wmma<<<dim3(Ndim / 64, OD / 64, NH), 128, 0, stream>>>(
        x, W0, h0, Ndim, FIN, OD);
    attn_coeff<<<(NH * Ndim) / 256, 256, 0, stream>>>(h0, as0, an0, fs0, fn0);
    gat_attention<<<dim3(Ndim / 16, NH), 128, 0, stream>>>(
        h0, fs0, fn0, adj, Mprox, out0, /*mode=*/0);

    // ---- layer 1 ----
    gemm_bf16_wmma<<<dim3(Ndim / 64, OD / 64, NH), 128, 0, stream>>>(
        out0, W1, h1, Ndim, HOD, OD);
    attn_coeff<<<(NH * Ndim) / 256, 256, 0, stream>>>(h1, as1, an1, fs1, fn1);
    gat_attention<<<dim3(Ndim / 16, NH), 128, 0, stream>>>(
        h1, fs1, fn1, adj, Mprox, yh1, /*mode=*/1);

    // ---- head mean + L2 normalize -> z ----
    mean_normalize<<<Ndim, 256, 0, stream>>>(yh1, zout, zbf);

    // ---- decode: sigmoid(z z^T) ----
    decode_sigmoid<<<dim3(Ndim / 64, Ndim / 16), 128, 0, stream>>>(zbf, out);
}
